// SelfAttention_51728586113658
// MI455X (gfx1250) — matrix operand order repaired and checked by reading, not compiled, not run on hardware
//
#include <hip/hip_runtime.h>
#include <hip/hip_bf16.h>
#include <stdint.h>

// ---------------- CDNA5 WMMA types ----------------
typedef __attribute__((ext_vector_type(16))) __bf16 v16bf;
typedef __attribute__((ext_vector_type(8)))  float  v8f;

union FragU { v16bf v; uint4 q[2]; };

// A-fragment (16x32 bf16, M=row per lane%16):
// lane<16: K=[0..15] ; lane>=16: K=[16..31]
__device__ __forceinline__ v16bf load_a_frag(const uint16_t* rowptr, int h) {
  FragU f;
  const char* p = (const char*)rowptr;
  f.q[0] = *(const uint4*)(p + h * 32);
  f.q[1] = *(const uint4*)(p + 16 + h * 32);
  return f.v;
}
// B-fragment (32x16 bf16, N=col per lane%16):
// lane<16: K=[0..15], lane>=16: K=[16..31]  (column of B = 64 contiguous bytes)
__device__ __forceinline__ v16bf load_b_frag(const uint16_t* colptr, int h) {
  FragU f;
  const char* p = (const char*)colptr + h * 32;
  f.q[0] = *(const uint4*)(p);
  f.q[1] = *(const uint4*)(p + 16);
  return f.v;
}

__device__ __forceinline__ uint32_t bf16rne(float x) {
  uint32_t u = __float_as_uint(x);
  return ((u + 0x7FFFu + ((u >> 16) & 1u)) >> 16) & 0xFFFFu;
}
__device__ __forceinline__ uint32_t bf16pack(float lo, float hi) {
  return bf16rne(lo) | (bf16rne(hi) << 16);
}

__device__ __forceinline__ v8f wmma_bf16(v16bf a, v16bf b, v8f c) {
  return __builtin_amdgcn_wmma_f32_16x16x32_bf16(false, a, false, b, (short)0, c,
                                                 false, false);
}

// Async global->LDS copy (ASYNCcnt), CDNA5 path. lds = low-32 generic offset.
__device__ __forceinline__ void async_b128(uint32_t lds, const void* g) {
  asm volatile("global_load_async_to_lds_b128 %0, %1, off"
               :: "v"(lds), "v"(g) : "memory");
}

#define NTOK 4096
#define CDIM 256
#define QKD  32
#define KSTRIDE 40  // LDS row stride in u16 elements (80 B, 16B-aligned, bank-skewed)

// LDS A-fragment load: same element layout as load_a_frag but with KSTRIDE rows.
__device__ __forceinline__ v16bf lds_a_frag(const uint16_t* base, int row, int h) {
  FragU f;
  const char* p = (const char*)(base + row * KSTRIDE);
  f.q[0] = *(const uint4*)(p + h * 32);
  f.q[1] = *(const uint4*)(p + 16 + h * 32);
  return f.v;
}

// ---------------- Kernel 1: pack wq/wk/wv into one bf16 matrix Wb[320][256] ----
__global__ void pack_weights_kernel(const float* wq, const float* wk,
                                    const float* wv, uint16_t* Wb) {
  int i = blockIdx.x * blockDim.x + threadIdx.x;
  if (i >= 320 * CDIM) return;
  int row = i >> 8, col = i & 255;
  float v;
  if (row < 32)      v = wq[row * CDIM + col];
  else if (row < 64) v = wk[(row - 32) * CDIM + col];
  else               v = wv[(row - 64) * CDIM + col];
  Wb[i] = (uint16_t)bf16rne(v);
}

// ---------------- Kernel 2: x[b][c][n] f32 -> xt[b][n][c] bf16 (LDS transpose) -
__global__ void transpose_kernel(const float* x, uint16_t* xt) {
  __shared__ uint16_t tile[32][33];
  const int nb = blockIdx.x * 32, cb = blockIdx.y * 32, b = blockIdx.z;
  const float* xp = x + ((size_t)b * CDIM + cb) * NTOK + nb;
#pragma unroll
  for (int i = 0; i < 4; i++) {
    int c = threadIdx.y + i * 8;
    tile[threadIdx.x][c] = (uint16_t)bf16rne(xp[(size_t)c * NTOK + threadIdx.x]);
  }
  __syncthreads();
#pragma unroll
  for (int i = 0; i < 4; i++) {
    int n = threadIdx.y + i * 8;
    xt[((size_t)b * NTOK + nb + n) * CDIM + cb + threadIdx.x] = tile[n][threadIdx.x];
  }
}

// ---------------- Kernel 3: projections via WMMA --------------------------------
// Per wave: one 16-token tile. Computes Qb[n][32], Kb[n][32], Vt[c][n] (bf16).
__global__ void __launch_bounds__(256) proj_kernel(
    const uint16_t* xt, const uint16_t* Wb, const float* bq, const float* bk,
    const float* bv, uint16_t* Qb, uint16_t* Kb, uint16_t* Vt) {
  const int lane = threadIdx.x & 31;
  const int h = lane >> 4, l16 = lane & 15;
  const int wid = blockIdx.x * (blockDim.x >> 5) + (threadIdx.x >> 5);
  const int b = wid >> 8;
  const int nb = (wid & 255) << 4;
  const uint16_t* xbase = xt + (size_t)b * NTOK * CDIM;

  v8f accq[2], acck[2], accv[16];
#pragma unroll
  for (int t = 0; t < 2; t++) accq[t] = acck[t] = (v8f){0, 0, 0, 0, 0, 0, 0, 0};
#pragma unroll
  for (int t = 0; t < 16; t++) accv[t] = (v8f){0, 0, 0, 0, 0, 0, 0, 0};

  for (int cb = 0; cb < CDIM; cb += 32) {
    v16bf xa = load_a_frag(xbase + (size_t)(nb + l16) * CDIM + cb, h); // M=n
    v16bf xb = load_b_frag(xbase + (size_t)(nb + l16) * CDIM + cb, h); // N=n
#pragma unroll
    for (int t = 0; t < 2; t++) {
      v16bf wqf = load_b_frag(Wb + (size_t)(t * 16 + l16) * CDIM + cb, h);
      accq[t] = wmma_bf16(xa, wqf, accq[t]);
      v16bf wkf = load_b_frag(Wb + (size_t)(32 + t * 16 + l16) * CDIM + cb, h);
      acck[t] = wmma_bf16(xa, wkf, acck[t]);
    }
#pragma unroll
    for (int ct = 0; ct < 16; ct++) {
      v16bf wvf = load_a_frag(Wb + (size_t)(64 + ct * 16 + l16) * CDIM + cb, h);
      accv[ct] = wmma_bf16(wvf, xb, accv[ct]);
    }
  }
  // Q/K: D tile M=n (reg r -> n = nb+r+8h), N=o (lane%16)
#pragma unroll
  for (int t = 0; t < 2; t++) {
    float biq = bq[t * 16 + l16];
    float bik = bk[t * 16 + l16];
#pragma unroll
    for (int r = 0; r < 8; r++) {
      int n = nb + r + 8 * h;
      size_t base = ((size_t)b * NTOK + n) * QKD + t * 16 + l16;
      Qb[base] = (uint16_t)bf16rne(accq[t][r] + biq);
      Kb[base] = (uint16_t)bf16rne(acck[t][r] + bik);
    }
  }
  // V: D tile M=c (reg), N=n (lane%16) -> Vt[b][c][n]
#pragma unroll
  for (int ct = 0; ct < 16; ct++) {
#pragma unroll
    for (int r = 0; r < 8; r++) {
      int c = ct * 16 + r + 8 * h;
      Vt[((size_t)b * CDIM + c) * NTOK + nb + l16] =
          (uint16_t)bf16rne(accv[ct][r] + bv[c]);
    }
  }
}

// ---------------- Kernel 4: fused flash attention, async-LDS staged K/V ---------
// WG = 8 waves = 128 queries. Per key-block of 32: WG cooperatively async-copies
// K tile (32x32) + V tile (256x32) to LDS (double-buffered, ASYNCcnt-tracked),
// then each wave computes S^T = K*Q^T (softmax-over-keys = in-lane reduce +
// shfl_xor(16)) and O^T += V*P with WMMA fragments read from LDS.
__global__ void __launch_bounds__(256) attn_kernel(
    const uint16_t* Qb, const uint16_t* Kb, const uint16_t* Vt, const float* x,
    const float* gamma, float* out) {
  __shared__ __align__(16) uint16_t Ks[2][32 * KSTRIDE];
  __shared__ __align__(16) uint16_t Vs[2][256 * KSTRIDE];

  const int tid = threadIdx.x;
  const int lane = tid & 31;
  const int h = lane >> 4, l16 = lane & 15;
  const int w = tid >> 5;
  const int b = blockIdx.x >> 5;              // 32 query-blocks per batch
  const int qb = ((blockIdx.x & 31) << 7) + w * 16;

  const uint16_t* Qbase = Qb + (size_t)b * NTOK * QKD;
  const uint16_t* Kbase = Kb + (size_t)b * NTOK * QKD;
  const uint16_t* Vbase = Vt + (size_t)b * CDIM * NTOK;

  // Stage kb-block into LDS buffer `buf`. Exactly 5 async instructions per wave:
  // 4 chunks of this thread's V row + 1 K chunk (threads 128..255 duplicate K
  // chunks so every wave's ASYNCcnt increments uniformly).
  auto stage = [&](int kb, int buf) {
    const char* vsrc = (const char*)(Vbase + (size_t)tid * NTOK + kb);
    uint32_t vdst = (uint32_t)(uintptr_t)&Vs[buf][tid * KSTRIDE];
#pragma unroll
    for (int j = 0; j < 4; j++) async_b128(vdst + j * 16, vsrc + j * 16);
    int id = tid & 127;
    int kr = id >> 2, kj = id & 3;
    const char* ksrc = (const char*)(Kbase + (size_t)(kb + kr) * QKD) + kj * 16;
    async_b128((uint32_t)(uintptr_t)&Ks[buf][kr * KSTRIDE] + kj * 16, ksrc);
  };

  stage(0, 0);

  v16bf qfrag = load_b_frag(Qbase + (size_t)(qb + l16) * QKD, h); // B: K=dim,N=query

  v8f o_acc[16];
#pragma unroll
  for (int t = 0; t < 16; t++) o_acc[t] = (v8f){0, 0, 0, 0, 0, 0, 0, 0};
  float m = -3.0e38f, l = 0.0f;
  const v8f zero = (v8f){0, 0, 0, 0, 0, 0, 0, 0};

  for (int it = 0; it < NTOK / 32; ++it) {
    const int cur = it & 1;
    if (it + 1 < NTOK / 32) {
      stage((it + 1) * 32, cur ^ 1);
      asm volatile("s_wait_asynccnt 5" ::: "memory"); // only the 5 new in flight
    } else {
      asm volatile("s_wait_asynccnt 0" ::: "memory");
    }
    __syncthreads(); // all waves' staged data for `cur` visible

    v16bf ka0 = lds_a_frag(&Ks[cur][0], l16, h);      // keys kb..kb+15
    v16bf ka1 = lds_a_frag(&Ks[cur][0], 16 + l16, h); // keys kb+16..kb+31
    v8f s0 = wmma_bf16(ka0, qfrag, zero); // S^T: M=key, N=query(lane)
    v8f s1 = wmma_bf16(ka1, qfrag, zero);

    float bm = s0[0];
#pragma unroll
    for (int i = 1; i < 8; i++) bm = fmaxf(bm, s0[i]);
#pragma unroll
    for (int i = 0; i < 8; i++) bm = fmaxf(bm, s1[i]);
    bm = fmaxf(bm, __shfl_xor(bm, 16, 32));
    float mnew = fmaxf(m, bm);
    float alpha = __expf(m - mnew);

    v8f p0, p1;
    float rs = 0.0f;
#pragma unroll
    for (int i = 0; i < 8; i++) { p0[i] = __expf(s0[i] - mnew); rs += p0[i]; }
#pragma unroll
    for (int i = 0; i < 8; i++) { p1[i] = __expf(s1[i] - mnew); rs += p1[i]; }
    rs += __shfl_xor(rs, 16, 32);
    l = l * alpha + rs;
    m = mnew;

#pragma unroll
    for (int t = 0; t < 16; t++)
#pragma unroll
      for (int i = 0; i < 8; i++) o_acc[t][i] *= alpha;

    // P^T (D-layout) -> P B-fragment (K=key 0..31, N=query)
    uint32_t a0 = bf16pack(p0[0], p0[1]), a1 = bf16pack(p0[2], p0[3]);
    uint32_t a2 = bf16pack(p0[4], p0[5]), a3 = bf16pack(p0[6], p0[7]);
    uint32_t b0 = bf16pack(p1[0], p1[1]), b1 = bf16pack(p1[2], p1[3]);
    uint32_t b2 = bf16pack(p1[4], p1[5]), b3 = bf16pack(p1[6], p1[7]);
    uint32_t ap0 = __shfl_xor((int)a0, 16, 32), ap1 = __shfl_xor((int)a1, 16, 32);
    uint32_t ap2 = __shfl_xor((int)a2, 16, 32), ap3 = __shfl_xor((int)a3, 16, 32);
    uint32_t bp0 = __shfl_xor((int)b0, 16, 32), bp1 = __shfl_xor((int)b1, 16, 32);
    uint32_t bp2 = __shfl_xor((int)b2, 16, 32), bp3 = __shfl_xor((int)b3, 16, 32);
    FragU pf;
    pf.q[0] = h ? make_uint4(bp0, bp1, bp2, bp3) : make_uint4(a0, a1, a2, a3);
    pf.q[1] = h ? make_uint4(b0, b1, b2, b3) : make_uint4(ap0, ap1, ap2, ap3);

    // O^T[c][q] += V^T(A: M=channel,K=key from LDS) * P(B)
#pragma unroll
    for (int ct = 0; ct < 16; ct++) {
      v16bf va = lds_a_frag(&Vs[cur][0], ct * 16 + l16, h);
      o_acc[ct] = wmma_bf16(va, pf.v, o_acc[ct]);
    }
    __syncthreads(); // protect `cur` buffer before it is re-staged next iteration
  }

  float inv = 1.0f / l;
  float g = gamma[0];
#pragma unroll
  for (int ct = 0; ct < 16; ct++) {
#pragma unroll
    for (int r = 0; r < 8; r++) {
      int c = ct * 16 + r + 8 * h;
      size_t idx = ((size_t)b * CDIM + c) * NTOK + qb + l16;
      out[idx] = g * (o_acc[ct][r] * inv) + x[idx];
    }
  }
}

// ---------------- Host launch ----------------------------------------------------
extern "C" void kernel_launch(void* const* d_in, const int* in_sizes, int n_in,
                              void* d_out, int out_size, void* d_ws, size_t ws_size,
                              hipStream_t stream) {
  const float* x     = (const float*)d_in[0];
  const float* wq    = (const float*)d_in[1];
  const float* bq    = (const float*)d_in[2];
  const float* wk    = (const float*)d_in[3];
  const float* bk    = (const float*)d_in[4];
  const float* wv    = (const float*)d_in[5];
  const float* bv    = (const float*)d_in[6];
  const float* gamma = (const float*)d_in[7];
  float* out = (float*)d_out;

  char* ws = (char*)d_ws;
  const size_t OFF_WB = 0;                            // 320*256*2   = 160 KB
  const size_t OFF_XT = OFF_WB + 320 * 256 * 2;       // 4*4096*256*2 = 8 MB
  const size_t OFF_QB = OFF_XT + (size_t)4 * NTOK * CDIM * 2;
  const size_t OFF_KB = OFF_QB + (size_t)4 * NTOK * QKD * 2;
  const size_t OFF_VT = OFF_KB + (size_t)4 * NTOK * QKD * 2;
  uint16_t* Wb = (uint16_t*)(ws + OFF_WB);
  uint16_t* xt = (uint16_t*)(ws + OFF_XT);
  uint16_t* Qb = (uint16_t*)(ws + OFF_QB);
  uint16_t* Kb = (uint16_t*)(ws + OFF_KB);
  uint16_t* Vt = (uint16_t*)(ws + OFF_VT);

  pack_weights_kernel<<<(320 * 256 + 255) / 256, 256, 0, stream>>>(wq, wk, wv, Wb);
  transpose_kernel<<<dim3(NTOK / 32, CDIM / 32, 4), dim3(32, 8), 0, stream>>>(x, xt);
  // 4 batches * 256 token-tiles = 1024 waves, 8 waves/block
  proj_kernel<<<128, 256, 0, stream>>>(xt, Wb, bq, bk, bv, Qb, Kb, Vt);
  // 4 batches * 32 query-blocks (128 queries each)
  attn_kernel<<<128, 256, 0, stream>>>(Qb, Kb, Vt, x, gamma, out);
}
